// Model_84774064488748
// MI455X (gfx1250) — matrix operand order repaired and checked by reading, not compiled, not run on hardware
//
#include <hip/hip_runtime.h>
#include <hip/hip_bf16.h>

// Problem constants (match reference).
#define NC    200      // cells
#define NCPAD 208      // cells padded to 13 full 16-row WMMA tiles
#define NG    2000     // genes_oi
#define NGTOT 20000    // total genes
#define NK    128      // spline vertices
#define NF    300000   // fragments
#define LOG_GTOT 9.903487552536127f  // log(20000)

typedef __attribute__((ext_vector_type(16))) _Float16 v16h;
typedef __attribute__((ext_vector_type(8)))  float    v8f;

// ---------------------------------------------------------------------------
// Kernel 1: per-gene spline geometry.
// widths = softmax(uw[gg]), bin_locs = [0, cumsum(widths)] with last forced
// to 1.0, t = trapezoid weights so that area = sum_k h[k]*t[k].
// One block (128 threads) per local gene.
// ---------------------------------------------------------------------------
__global__ __launch_bounds__(128) void k_spline_prep(
    const float* __restrict__ uw, const int* __restrict__ genes_oi,
    float* __restrict__ widths, float* __restrict__ bin_locs,
    float* __restrict__ tw) {
  __shared__ float sh[128];
  __shared__ float sc[128];
  const int g = blockIdx.x;
  const int t = threadIdx.x;
  const int gg = genes_oi[g];

  float e = (t < NK - 1) ? uw[(size_t)gg * (NK - 1) + t] : -3.4e38f;
  // max-reduce
  sh[t] = e; __syncthreads();
  for (int off = 64; off > 0; off >>= 1) {
    if (t < off) sh[t] = fmaxf(sh[t], sh[t + off]);
    __syncthreads();
  }
  const float m = sh[0]; __syncthreads();
  const float ex = (t < NK - 1) ? expf(e - m) : 0.0f;
  sh[t] = ex; __syncthreads();
  for (int off = 64; off > 0; off >>= 1) {
    if (t < off) sh[t] += sh[t + off];
    __syncthreads();
  }
  const float Z = sh[0]; __syncthreads();
  const float w = ex / Z;  // width; 0 at slot 127

  // inclusive Hillis-Steele scan of w
  sc[t] = w; __syncthreads();
  for (int off = 1; off < 128; off <<= 1) {
    float v = (t >= off) ? sc[t - off] : 0.0f;
    __syncthreads();
    sc[t] += v;
    __syncthreads();
  }
  float bl = (t == 0) ? 0.0f : sc[t - 1];  // bin_locs[t] = sum(w[0..t-1])
  if (t == NK - 1) bl = 1.0f;              // reference forces last edge to 1
  bin_locs[(size_t)g * NK + t] = bl;
  if (t < NK - 1) widths[(size_t)g * (NK - 1) + t] = w;

  // trapezoid weights
  sh[t] = w; __syncthreads();  // sh[127] == 0
  float tv;
  if (t == 0)            tv = 0.5f * sh[0];
  else if (t == NK - 1)  tv = 0.5f * sh[NK - 2];
  else                   tv = 0.5f * (sh[t - 1] + sh[t]);
  tw[(size_t)g * NK + t] = tv;
}

// ---------------------------------------------------------------------------
// Kernel 2: per-cell log-sum-exp of overall logits over all GTOT genes.
// One block (256 threads) per cell; streaming online logsumexp.
// ---------------------------------------------------------------------------
__global__ __launch_bounds__(256) void k_logz(
    const float* __restrict__ latent, const float* __restrict__ oslope,
    const float* __restrict__ obase, float* __restrict__ logZ) {
  const int c = blockIdx.x;
  const int t = threadIdx.x;
  const float lat = latent[c];
  float m = -3.4e38f, s = 0.0f;
  for (int j = t; j < NGTOT; j += 256) {
    const float v = obase[j] + lat * oslope[j];
    if (v <= m) {
      s += expf(v - m);
    } else {
      s = s * expf(m - v) + 1.0f;
      m = v;
    }
  }
  __shared__ float sm[256], ss[256];
  sm[t] = m; ss[t] = s; __syncthreads();
  for (int off = 128; off > 0; off >>= 1) {
    if (t < off) {
      const float m2 = sm[t + off], s2 = ss[t + off];
      const float M = fmaxf(sm[t], m2);
      ss[t] = ss[t] * expf(sm[t] - M) + s2 * expf(m2 - M);
      sm[t] = M;
    }
    __syncthreads();
  }
  if (t == 0) logZ[c] = sm[0] + logf(ss[0]);
}

// ---------------------------------------------------------------------------
// Kernel 3: log_area(c,g) table via WMMA trapezoid reduction.
// One block (128 thr = 4 waves) per local gene. Each wave owns cell-tiles of
// 16; per 32-wide K chunk it builds A = exp(a + lat*s) in the documented
// 16-bit 16x32 A layout and B = trapezoid weights replicated across N, then
// accumulates with v_wmma_f32_16x16x32_f16. Column N=0 of the accumulator
// (lanes 0 and 16) holds the 16 areas. log_area is padded to NCPAD cell rows
// so the stores are unconditional (no post-WMMA exec-mask ladder).
// ---------------------------------------------------------------------------
__global__ __launch_bounds__(128) void k_area(
    const float* __restrict__ uh, const float* __restrict__ hsl,
    const int* __restrict__ genes_oi, const float* __restrict__ latent,
    const float* __restrict__ tw, float* __restrict__ log_area) {
  __shared__ float sa[NK], ss_[NK], st[NK], slat[NCPAD];
  const int g = blockIdx.x;
  const int t = threadIdx.x;
  const int gg = genes_oi[g];
  const size_t base = (size_t)gg * NK;
  sa[t]  = uh[base + t];
  ss_[t] = hsl[base + t];
  st[t]  = tw[(size_t)g * NK + t];
  for (int i = t; i < NCPAD; i += 128) slat[i] = (i < NC) ? latent[i] : 0.0f;
  __syncthreads();

  const int wave = t >> 5;
  const int lane = t & 31;
  const int half = lane >> 4;   // 0: lanes 0-15, 1: lanes 16-31
  const int mrow = lane & 15;   // cell row within tile (A-matrix M)
  const int ntiles = NCPAD / 16;  // 13

  for (int tile = wave; tile < ntiles; tile += 4) {
    const float lat = slat[tile * 16 + mrow];
    v8f acc = {};
    for (int chunk = 0; chunk < 4; ++chunk) {
      const int kb = chunk * 32;
      v16h A, B;
#pragma unroll
      for (int e = 0; e < 16; ++e) {
        // A (16-bit 16x32): lanes 0-15 hold K = 0-7,16-23; lanes 16-31 hold
        // K = 8-15,24-31 (offset +8 per half).
        const int k = kb + ((e < 8) ? e : e + 8) + (half ? 8 : 0);
        const float hv = __expf(sa[k] + lat * ss_[k]);  // feeds f16: fast exp
        A[e] = (_Float16)hv;
        // B (16-bit 32x16): same K order as A (0-7,16-23 / 8-15,24-31); same weights
        // for every N column -> all D columns equal the area.
        B[e] = (_Float16)st[kb + ((e < 8) ? e : e + 8) + 8 * half];
      }
      acc = __builtin_amdgcn_wmma_f32_16x16x32_f16(
          false, A, false, B, (short)0, acc, false, false);
    }
    // C/D layout: lane n<16 -> (M=r, N=n); lane>=16 -> (M=8+r, N=lane-16).
    // N==0 columns live in lanes 0 and 16; store all 16 rows unconditionally
    // into the padded table.
    if (mrow == 0) {
#pragma unroll
      for (int r = 0; r < 8; ++r) {
        const int cell = tile * 16 + r + 8 * half;
        log_area[(size_t)cell * NG + g] = logf(acc[r]);
      }
    }
  }
}

// ---------------------------------------------------------------------------
// Kernel 4: per-fragment likelihood. Binary search the gene's bin edges,
// 2 exps for the bin-end heights, subtract log_area and add the overall
// log-softmax term. Block-level tree reduction -> partials.
// ---------------------------------------------------------------------------
__global__ __launch_bounds__(256) void k_frag(
    const float* __restrict__ latent, const float* __restrict__ hsl,
    const float* __restrict__ oslope, const float* __restrict__ obase,
    const float* __restrict__ uh, const float* __restrict__ coords,
    const int* __restrict__ genes_oi, const int* __restrict__ cxg,
    const int* __restrict__ glocal,
    const float* __restrict__ widths, const float* __restrict__ bin_locs,
    const float* __restrict__ log_area, const float* __restrict__ logZ,
    float* __restrict__ partials) {
  const int f = blockIdx.x * 256 + threadIdx.x;
  float ll = 0.0f;
  if (f < NF) {
    const int gl   = glocal[f];
    const int cg   = cxg[f];
    const int cell = cg / NG;          // cxg = cell*NG + gl
    const int gg   = genes_oi[gl];
    const float x  = coords[f];

    // bin index = clip(#edges(1..127) <= x, 0, 126); edge[127]==1.0 > x.
    const float* arr = bin_locs + (size_t)gl * NK;
    int lo = 1, hi = NK - 1;
    while (lo < hi) {
      const int mid = (lo + hi) >> 1;
      if (arr[mid] <= x) lo = mid + 1; else hi = mid;
    }
    int b = lo - 1;
    if (b > NK - 2) b = NK - 2;        // clamp to 126
    const float left  = arr[b];
    const float w     = widths[(size_t)gl * (NK - 1) + b];
    const float alpha = (x - left) / w;

    const float lat = latent[cell];
    const size_t hb = (size_t)gg * NK + b;
    const float hl = expf(uh[hb]     + lat * hsl[hb]);
    const float hr = expf(uh[hb + 1] + lat * hsl[hb + 1]);
    const float pdf = hl + alpha * (hr - hl);
    const float height_ll = logf(pdf) - log_area[(size_t)cell * NG + gl];

    const float logit = obase[gg] + lat * oslope[gg];
    const float overall_ll = logit - logZ[cell] + LOG_GTOT;
    ll = height_ll + overall_ll;
  }
  __shared__ float sh[256];
  sh[threadIdx.x] = ll; __syncthreads();
  for (int off = 128; off > 0; off >>= 1) {
    if (threadIdx.x < off) sh[threadIdx.x] += sh[threadIdx.x + off];
    __syncthreads();
  }
  if (threadIdx.x == 0) partials[blockIdx.x] = sh[0];
}

// ---------------------------------------------------------------------------
// Kernel 5: deterministic final reduction; writes d_out[0] = -sum.
// ---------------------------------------------------------------------------
__global__ __launch_bounds__(256) void k_final(
    const float* __restrict__ partials, int n, float* __restrict__ out) {
  __shared__ float sh[256];
  float s = 0.0f;
  for (int i = threadIdx.x; i < n; i += 256) s += partials[i];
  sh[threadIdx.x] = s; __syncthreads();
  for (int off = 128; off > 0; off >>= 1) {
    if (threadIdx.x < off) sh[threadIdx.x] += sh[threadIdx.x + off];
    __syncthreads();
  }
  if (threadIdx.x == 0) out[0] = -sh[0];
}

// ---------------------------------------------------------------------------
extern "C" void kernel_launch(void* const* d_in, const int* in_sizes, int n_in,
                              void* d_out, int out_size, void* d_ws, size_t ws_size,
                              hipStream_t stream) {
  const float* latent = (const float*)d_in[0];   // (C,1)
  const float* hsl    = (const float*)d_in[1];   // (GTOT,K) height_slope
  const float* oslope = (const float*)d_in[2];   // (GTOT,1)
  const float* obase  = (const float*)d_in[3];   // (GTOT,)
  const float* uh     = (const float*)d_in[4];   // (GTOT,K)
  const float* uw     = (const float*)d_in[5];   // (GTOT,K-1)
  const float* coords = (const float*)d_in[6];   // (F,)
  const int*   goi    = (const int*)d_in[7];     // (G,)
  const int*   cxg    = (const int*)d_in[8];     // (F,) cell*G + glocal
  // d_in[9] (cut_localcellxgene_ix) reconstructed from cxg + genes_oi
  const int*   glocal = (const int*)d_in[10];    // (F,)
  float* out = (float*)d_out;

  // Workspace carve-up (floats)
  float* ws = (float*)d_ws;
  float* w_widths  = ws;                                    // G*(K-1) = 254000
  float* w_binlocs = w_widths  + (size_t)NG * (NK - 1);     // G*K     = 256000
  float* w_tw      = w_binlocs + (size_t)NG * NK;           // G*K     = 256000
  float* w_logarea = w_tw      + (size_t)NG * NK;           // NCPAD*G = 416000
  float* w_logz    = w_logarea + (size_t)NCPAD * NG;        // C       = 200
  float* w_part    = w_logz    + NC;                        // frag blocks

  const int nfragblk = (NF + 255) / 256;

  k_spline_prep<<<NG, 128, 0, stream>>>(uw, goi, w_widths, w_binlocs, w_tw);
  k_logz<<<NC, 256, 0, stream>>>(latent, oslope, obase, w_logz);
  k_area<<<NG, 128, 0, stream>>>(uh, hsl, goi, latent, w_tw, w_logarea);
  k_frag<<<nfragblk, 256, 0, stream>>>(latent, hsl, oslope, obase, uh, coords,
                                       goi, cxg, glocal, w_widths, w_binlocs,
                                       w_logarea, w_logz, w_part);
  k_final<<<1, 256, 0, stream>>>(w_part, nfragblk, out);
}
